// Attention_37520834297905
// MI455X (gfx1250) — matrix operand order repaired and checked by reading, not compiled, not run on hardware
//
#include <hip/hip_runtime.h>
#include <hip/hip_fp16.h>

typedef __attribute__((ext_vector_type(16))) _Float16 v16h;
typedef __attribute__((ext_vector_type(8)))  _Float16 v8h;
typedef __attribute__((ext_vector_type(8)))  float    v8f;

#define WSZ    8
#define SHF    4
#define HEADS  6
#define DIM    192
#define IMG    256

// LDS row strides (halves); all chosen so 8-half vectors are 16B-aligned
#define LDX    200   // s_xT / s_outT: [64 pixels][192 ch + pad]
#define LDT    104   // s_tT: [64 pixels][96 ch + pad]
#define LDQ    72    // s_qk: [64 rows (q|k)][64 px + pad]
#define LDV    40    // s_vT: [64 pixels][32 d + pad]
#define LDA    40    // s_ah: [32 c][32 d + pad]

// ---- WMMA fragment helpers (gfx1250 wave32 layouts, CDNA5 ISA) ----

// A 16x32 f16, memory row-major [M][K], row stride ld. 16 contiguous halves (k = 16h + s).
static __device__ __forceinline__ v16h load_A_rm(const _Float16* tile, int ld) {
  const int lane = threadIdx.x & 31;
  const _Float16* r = tile + (lane & 15) * ld;
  const int kb = (lane < 16) ? 0 : 16;
  v16h a;
#pragma unroll
  for (int i = 0; i < 8; ++i) a[i] = r[kb + i];
#pragma unroll
  for (int i = 0; i < 8; ++i) a[8 + i] = r[kb + 8 + i];
  return a;
}

// B 32x16 f16 where memory holds element (k,n) at tile[n*ld + k]  ([N][K] row-major).
// Per lane: 16 contiguous halves -> 2x ds_load_b128.
static __device__ __forceinline__ v16h load_B_nk(const _Float16* tile, int ld) {
  const int lane = threadIdx.x & 31;
  const int kb   = (lane < 16) ? 0 : 16;
  const _Float16* r = tile + (lane & 15) * ld + kb;
  v16h b;
#pragma unroll
  for (int i = 0; i < 16; ++i) b[i] = r[i];
  return b;
}

static __device__ __forceinline__ v8f wmma16(v16h a, v16h b, v8f c) {
  return __builtin_amdgcn_wmma_f32_16x16x32_f16(false, a, false, b, (short)0, c,
                                                false, false);
}

// C 16x16 f32 -> f16 written TRANSPOSED: element (m,n) at baseT[n*ld + m].
// Per lane n is fixed, m contiguous -> single 16B LDS store.
static __device__ __forceinline__ void store_C_f16_T(v8f c, _Float16* baseT, int ld) {
  const int lane = threadIdx.x & 31;
  const int col  = lane & 15;
  const int mb   = (lane < 16) ? 0 : 8;
  v8h hv;
#pragma unroll
  for (int r = 0; r < 8; ++r) hv[r] = (_Float16)c[r];
  *(v8h*)(baseT + col * ld + mb) = hv;
}

// C 16x16 f32 row-major store (for the small attn logits tile)
static __device__ __forceinline__ void store_C_f32(v8f c, float* tile, int ld) {
  const int lane = threadIdx.x & 31;
  const int col  = lane & 15;
  const int mb   = (lane < 16) ? 0 : 8;
#pragma unroll
  for (int r = 0; r < 8; ++r) tile[(mb + r) * ld + col] = c[r];
}

// -------- weight f32 -> f16 conversion (one-shot, L2-resident afterwards) --------
__global__ void cvt_weights(const float* __restrict__ qw, const float* __restrict__ pw,
                            __half* __restrict__ qh, __half* __restrict__ ph) {
  int i = blockIdx.x * 256 + threadIdx.x;
  if (i < 3 * DIM * DIM) ((_Float16*)qh)[i] = (_Float16)qw[i];
  if (i < DIM * DIM)     ((_Float16*)ph)[i] = (_Float16)pw[i];
}

// -------- fused shifted-window transposed-attention kernel: 1 block per window --------
__global__ __launch_bounds__(256)
void swin_xca_kernel(const float* __restrict__ x,
                     const __half* __restrict__ qw_h,   // 576x192 f16
                     const float*  __restrict__ dw,     // 576x9 f32 depthwise
                     const __half* __restrict__ pw_h,   // 192x192 f16
                     const float*  __restrict__ temperature,
                     float* __restrict__ out) {
  extern __shared__ char smem_raw[];
  _Float16* s_xT   = (_Float16*)smem_raw;          // 64 x LDX : x window, [px][ch]
  _Float16* s_tT   = s_xT   + 64 * LDX;            // 64 x LDT : pre-dwconv qkv, [px][ch96]
  _Float16* s_qk   = s_tT   + 64 * LDT;            // 64 x LDQ : q|k rows, [ch][px]
  _Float16* s_vT   = s_qk   + 64 * LDQ;            // 64 x LDV : v, [px][d]
  _Float16* s_outT = s_vT   + 64 * LDV;            // 64 x LDX : attn out, [px][ch192]
  float*    s_attn = (float*)(s_outT + 64 * LDX);  // 32 x 33 f32 logits (alias s_dwT)
  _Float16* s_ah   = (_Float16*)(s_attn + 32 * 33);// 32 x LDA f16 softmaxed attn
  float*    s_dwT  = s_attn;                       // 9 x 96 staged dw weights [tap][ch]

  const _Float16* qwh = (const _Float16*)qw_h;
  const _Float16* pwh = (const _Float16*)pw_h;

  const int tid = threadIdx.x;
  const int wv  = tid >> 5;           // wave id 0..7
  const int nt  = wv & 3;             // this wave's fixed N-tile (pixel block)
  const int a2  = wv >> 2;            // 0..1, M-tile parity
  const int win = blockIdx.x;         // 4096 windows
  const int bi  = win >> 10;          // batch
  const int wy  = (win >> 5) & 31;
  const int wx  = win & 31;

  // ---- Phase 1: gather rolled window into LDS (transposed, f16) ----
  const float* xb = x + (size_t)bi * DIM * IMG * IMG;
  for (int e = tid; e < DIM * 64; e += 256) {
    int c = e >> 6, p = e & 63;
    int i = p >> 3, j = p & 7;
    int sr = (wy * WSZ + i + SHF) & (IMG - 1);   // roll(-SHF): rolled[r] = x[r+SHF]
    int sc = (wx * WSZ + j + SHF) & (IMG - 1);
    s_xT[p * LDX + c] = (_Float16)xb[((size_t)c * IMG + sr) * IMG + sc];
  }
  __syncthreads();

  // ---- Phase 2: per-head pipeline ----
  for (int h = 0; h < HEADS; ++h) {
    // 2a. 1x1-conv GEMM: t(96x64) = W_head(96x192) * x(192x64)
    //     wave owns nt; mt = a2 + 2*t3 (t3=part). B loaded once per ks, reused 3x.
    {
      v8f acc0 = {}, acc1 = {}, acc2 = {};
#pragma unroll
      for (int ks = 0; ks < 6; ++ks) {
        v16h b = load_B_nk(s_xT + (nt * 16) * LDX + ks * 32, LDX);
#pragma unroll
        for (int t3 = 0; t3 < 3; ++t3) {
          // mt = a2 + 2*t3 ; part = t3 ; row-in-part offset = a2*16
          const _Float16* abase =
              qwh + (size_t)(t3 * DIM + h * 32 + a2 * 16) * DIM + ks * 32;
          v16h a = load_A_rm(abase, DIM);
          if (t3 == 0) acc0 = wmma16(a, b, acc0);
          else if (t3 == 1) acc1 = wmma16(a, b, acc1);
          else acc2 = wmma16(a, b, acc2);
        }
      }
      store_C_f16_T(acc0, s_tT + (nt * 16) * LDT + (a2 + 0) * 16, LDT);
      store_C_f16_T(acc1, s_tT + (nt * 16) * LDT + (a2 + 2) * 16, LDT);
      store_C_f16_T(acc2, s_tT + (nt * 16) * LDT + (a2 + 4) * 16, LDT);
    }
    // stage this head's depthwise weights transposed: s_dwT[tap][ch96]
    for (int e = tid; e < 9 * 96; e += 256) {
      int tap = e / 96, lr = e - tap * 96;
      int gch = (lr >> 5) * DIM + h * 32 + (lr & 31);
      s_dwT[tap * 96 + lr] = dw[gch * 9 + tap];
    }
    __syncthreads();

    // 2b. depthwise 3x3 in-window (zero pad); 8 channels per task, vector LDS reads
    for (int t = 0; t < 3; ++t) {
      int task = tid + t * 256;             // 768 tasks = 64 px * 12 ch-groups
      int p = task / 12, g = task - p * 12;
      int i = p >> 3, j = p & 7;
      float acc[8];
#pragma unroll
      for (int u = 0; u < 8; ++u) acc[u] = 0.f;
#pragma unroll
      for (int ky = 0; ky < 3; ++ky) {
        int yy = i + ky - 1;
        if ((unsigned)yy < 8u) {
#pragma unroll
          for (int kx = 0; kx < 3; ++kx) {
            int xx = j + kx - 1;
            if ((unsigned)xx < 8u) {
              v8h in = *(const v8h*)(s_tT + (yy * 8 + xx) * LDT + g * 8);
              const float* w8 = s_dwT + (ky * 3 + kx) * 96 + g * 8;
#pragma unroll
              for (int u = 0; u < 8; ++u) acc[u] += w8[u] * (float)in[u];
            }
          }
        }
      }
      if (g < 8) {                 // q,k -> row-major [ch][px] for attn A + l2norm
#pragma unroll
        for (int u = 0; u < 8; ++u) s_qk[(g * 8 + u) * LDQ + p] = (_Float16)acc[u];
      } else {                     // v -> transposed [px][d] for PV GEMM B
        v8h hv;
#pragma unroll
        for (int u = 0; u < 8; ++u) hv[u] = (_Float16)acc[u];
        *(v8h*)(s_vT + p * LDV + (g - 8) * 8) = hv;
      }
    }
    __syncthreads();

    // 2c. l2-normalize q rows (fold temperature) and k rows; 4 threads per row
    {
      int row = tid >> 2, q4 = tid & 3;
      _Float16* rp = s_qk + row * LDQ + q4 * 16;
      float ss = 0.f;
#pragma unroll
      for (int u = 0; u < 16; ++u) { float v = (float)rp[u]; ss += v * v; }
      ss += __shfl_xor(ss, 1);
      ss += __shfl_xor(ss, 2);
      float sc = 1.f / fmaxf(sqrtf(ss), 1e-12f);
      if (row < 32) sc *= temperature[h];
#pragma unroll
      for (int u = 0; u < 16; ++u) rp[u] = (_Float16)((float)rp[u] * sc);
    }
    __syncthreads();

    // 2d. attn(32x32) = q(32x64) . k^T : A = q rows [M][K], B(k=n,N=d) from k rows [N][K]
    if (wv < 4) {
      const int mt = wv >> 1, ntt = wv & 1;
      v8f acc = {};
#pragma unroll
      for (int ks = 0; ks < 2; ++ks) {
        v16h a = load_A_rm(s_qk + (mt * 16) * LDQ + ks * 32, LDQ);
        v16h b = load_B_nk(s_qk + (32 + ntt * 16) * LDQ + ks * 32, LDQ);
        acc = wmma16(a, b, acc);
      }
      store_C_f32(acc, s_attn + (mt * 16) * 33 + ntt * 16, 33);
    }
    __syncthreads();

    // 2e. softmax over d; 8 threads per row
    {
      int row = tid >> 3, o8 = tid & 7;
      const float* rp = s_attn + row * 33 + o8 * 4;
      float m = fmaxf(fmaxf(rp[0], rp[1]), fmaxf(rp[2], rp[3]));
      m = fmaxf(m, __shfl_xor(m, 1));
      m = fmaxf(m, __shfl_xor(m, 2));
      m = fmaxf(m, __shfl_xor(m, 4));
      float e0 = __expf(rp[0] - m), e1 = __expf(rp[1] - m);
      float e2 = __expf(rp[2] - m), e3 = __expf(rp[3] - m);
      float s = e0 + e1 + e2 + e3;
      s += __shfl_xor(s, 1);
      s += __shfl_xor(s, 2);
      s += __shfl_xor(s, 4);
      float inv = 1.f / s;
      _Float16* op = s_ah + row * LDA + o8 * 4;
      op[0] = (_Float16)(e0 * inv); op[1] = (_Float16)(e1 * inv);
      op[2] = (_Float16)(e2 * inv); op[3] = (_Float16)(e3 * inv);
    }
    __syncthreads();

    // 2f. out(32x64) = attn(32x32) @ v(32x64); one tile per wave; C stored transposed
    {
      const int mt = wv >> 2;              // reuse nt = wv & 3
      v16h a = load_A_rm(s_ah + (mt * 16) * LDA, LDA);
      v16h b = load_B_nk(s_vT + (nt * 16) * LDV, LDV);
      v8f acc = {};
      acc = wmma16(a, b, acc);
      store_C_f16_T(acc, s_outT + (nt * 16) * LDX + h * 32 + mt * 16, LDX);
    }
    __syncthreads();
  }

  // ---- Phase 3: projection GEMM (192x192)*(192x64), rolled store ----
  // wave owns nt; mt = a2 + 2*t6; B loaded once per ks, reused 6x.
  float* ob = out + (size_t)bi * DIM * IMG * IMG;
  {
    v8f acc[6];
#pragma unroll
    for (int t6 = 0; t6 < 6; ++t6) acc[t6] = (v8f){};
#pragma unroll
    for (int ks = 0; ks < 6; ++ks) {
      v16h b = load_B_nk(s_outT + (nt * 16) * LDX + ks * 32, LDX);
#pragma unroll
      for (int t6 = 0; t6 < 6; ++t6) {
        const int mt = a2 + 2 * t6;
        v16h a = load_A_rm(pwh + (size_t)(mt * 16) * DIM + ks * 32, DIM);
        acc[t6] = wmma16(a, b, acc[t6]);
      }
    }
    const int lane = tid & 31;
    const int col  = lane & 15;
    const int mb   = (lane < 16) ? 0 : 8;
    const int p = nt * 16 + col;
    const int i = p >> 3, j = p & 7;
    const int dr = (wy * WSZ + i + SHF) & (IMG - 1);  // roll(+SHF) of rolled frame
    const int dc = (wx * WSZ + j + SHF) & (IMG - 1);
#pragma unroll
    for (int t6 = 0; t6 < 6; ++t6) {
      const int mt = a2 + 2 * t6;
#pragma unroll
      for (int r = 0; r < 8; ++r) {
        const int o = mt * 16 + mb + r;
        ob[((size_t)o * IMG + dr) * IMG + dc] = acc[t6][r];
      }
    }
  }
}

extern "C" void kernel_launch(void* const* d_in, const int* in_sizes, int n_in,
                              void* d_out, int out_size, void* d_ws, size_t ws_size,
                              hipStream_t stream) {
  (void)in_sizes; (void)n_in; (void)out_size; (void)ws_size;
  const float* x    = (const float*)d_in[0];
  const float* qw   = (const float*)d_in[1];   // (576,192)
  const float* dw   = (const float*)d_in[2];   // (576,1,3,3)
  const float* pw   = (const float*)d_in[3];   // (192,192)
  const float* temp = (const float*)d_in[4];   // (6,1,1)
  float* out = (float*)d_out;

  __half* qh = (__half*)d_ws;                  // 576*192 f16
  __half* ph = qh + 3 * DIM * DIM;             // 192*192 f16

  cvt_weights<<<(3 * DIM * DIM + 255) / 256, 256, 0, stream>>>(qw, pw, qh, ph);

  const size_t smem =
      (size_t)(64 * LDX + 64 * LDT + 64 * LDQ + 64 * LDV + 64 * LDX) * 2  // halves
      + 32 * 33 * sizeof(float)            // attn logits (aliases dw stage)
      + 32 * LDA * 2;                      // softmaxed attn
  swin_xca_kernel<<<4096, 256, smem, stream>>>(x, qh, dw, ph, temp, out);
}
